// Qwen3MoEMLP_2044404433452
// MI455X (gfx1250) — matrix operand order repaired and checked by reading, not compiled, not run on hardware
//
#include <hip/hip_runtime.h>
#include <math.h>

// Problem dims (fixed by reference)
#define HDIM   2048
#define IDIM   768
#define ENUM   8
#define TNUM   2048   // 2 * 1024 tokens
#define TM     32     // token-tile (M) per block: two 16-row WMMA tiles

typedef __attribute__((ext_vector_type(16))) __bf16          v16bf;
typedef __attribute__((ext_vector_type(8)))  float           v8f;
typedef __attribute__((ext_vector_type(8)))  unsigned short  v8us;

union BFPack { v16bf v; v8us h[2]; };

// fp32 -> bf16 RNE via native cast (gfx1250 has hw bf16 converts)
__device__ __forceinline__ unsigned short f2bf(float f) {
  __bf16 b = (__bf16)f;
  return __builtin_bit_cast(unsigned short, b);
}

// convert 8 consecutive fp32 -> 8 bf16 (two b128 loads)
__device__ __forceinline__ v8us cvt8(const float* __restrict__ p) {
  float4 a = *(const float4*)p;
  float4 b = *(const float4*)(p + 4);
  v8us d;
  d[0]=f2bf(a.x); d[1]=f2bf(a.y); d[2]=f2bf(a.z); d[3]=f2bf(a.w);
  d[4]=f2bf(b.x); d[5]=f2bf(b.y); d[6]=f2bf(b.z); d[7]=f2bf(b.w);
  return d;
}

// B operand, fallback path: 16 consecutive fp32 of a weight row -> v16bf
__device__ __forceinline__ v16bf cvt16(const float* __restrict__ p) {
  BFPack t; t.h[0] = cvt8(p); t.h[1] = cvt8(p + 8); return t.v;
}

// B operand, fast path: 16 consecutive bf16 (32 bytes, two b128 loads)
__device__ __forceinline__ v16bf ld16bf(const unsigned short* __restrict__ p) {
  BFPack t; t.h[0] = *(const v8us*)p; t.h[1] = *(const v8us*)(p + 8); return t.v;
}

// A operand from LDS: two 8-elem chunks per lane, contiguous K (matches B):
// K = kb..kb+7 and K = kb+8..kb+15, kb = (lane>=16) ? 16 : 0
__device__ __forceinline__ v16bf mk_lds(const unsigned short* lo,
                                        const unsigned short* hi) {
  BFPack t;
  t.h[0] = *(const v8us*)lo;
  t.h[1] = *(const v8us*)hi;
  return t.v;
}

__device__ __forceinline__ v8f wmma_bf16(v16bf a, v16bf b, v8f c) {
  // (neg_a, A, neg_b, B, c_mod, C, reuse_a, reuse_b)
  return __builtin_amdgcn_wmma_f32_16x16x32_bf16(false, a, false, b,
                                                 (short)0, c, false, false);
}

// CDNA5 async global->LDS copy (16B per lane), tracked by ASYNCcnt.
// VDST holds the wave-relative LDS byte offset = low 32 bits of the generic
// pointer.
__device__ __forceinline__ void async_copy_b128(unsigned lds_off,
                                                const void* gsrc) {
  asm volatile("global_load_async_to_lds_b128 %0, %1, off"
               :: "v"(lds_off), "v"(gsrc) : "memory");
}
__device__ __forceinline__ void wait_asynccnt0() {
  asm volatile("s_wait_asynccnt 0" ::: "memory");
}

// ---------------------------------------------------------------------------
// Kernel 0: zero output + expert counters
// ---------------------------------------------------------------------------
__global__ void zero_kernel(float4* __restrict__ out, int n4,
                            int* __restrict__ count) {
  int i = blockIdx.x * blockDim.x + threadIdx.x;
  int stride = gridDim.x * blockDim.x;
  float4 z = make_float4(0.f, 0.f, 0.f, 0.f);
  for (; i < n4; i += stride) out[i] = z;
  if (blockIdx.x == 0 && threadIdx.x < ENUM) count[threadIdx.x] = 0;
}

// ---------------------------------------------------------------------------
// Kernel 0b: bulk fp32 -> bf16 conversion (weights + activations). 151 MB of
// fp32 weights become 75 MB bf16 that then live in the 192 MB L2.
// ---------------------------------------------------------------------------
__global__ void cvt_kernel(const float* __restrict__ src,
                           unsigned short* __restrict__ dst, int n8) {
  int i = blockIdx.x * blockDim.x + threadIdx.x;
  int stride = gridDim.x * blockDim.x;
  for (; i < n8; i += stride) {
    v8us d = cvt8(src + (size_t)i * 8);
    *(v8us*)(dst + (size_t)i * 8) = d;
  }
}

// ---------------------------------------------------------------------------
// Kernel 1: router. One block per token; wave w computes logit for expert w.
// fp32 softmax, top-2 (first-occurrence ties like lax.top_k), renormalize,
// append (token, weight) to per-expert lists.
// ---------------------------------------------------------------------------
__global__ void router_kernel(const float* __restrict__ x,
                              const float* __restrict__ rw,
                              int* __restrict__ count,
                              int* __restrict__ tok_idx,
                              float* __restrict__ tok_w) {
  const int t    = blockIdx.x;
  const int wv   = threadIdx.x >> 5;
  const int lane = threadIdx.x & 31;
  __shared__ float s_logit[ENUM];

  const float* xr = x  + (size_t)t  * HDIM;
  const float* wr = rw + (size_t)wv * HDIM;
  float acc = 0.f;
  for (int h = lane; h < HDIM; h += 32) acc += xr[h] * wr[h];
#pragma unroll
  for (int off = 16; off > 0; off >>= 1) acc += __shfl_xor(acc, off, 32);
  if (lane == 0) s_logit[wv] = acc;
  __syncthreads();

  if (threadIdx.x == 0) {
    float p[ENUM];
    float m = s_logit[0];
    for (int e = 1; e < ENUM; e++) m = fmaxf(m, s_logit[e]);
    float s = 0.f;
    for (int e = 0; e < ENUM; e++) { p[e] = expf(s_logit[e] - m); s += p[e]; }
    float inv = 1.f / s;
    for (int e = 0; e < ENUM; e++) p[e] *= inv;

    int i1 = 0; float v1 = p[0];
    for (int e = 1; e < ENUM; e++) if (p[e] > v1) { v1 = p[e]; i1 = e; }
    int i2 = -1; float v2 = -1.f;
    for (int e = 0; e < ENUM; e++) if (e != i1 && p[e] > v2) { v2 = p[e]; i2 = e; }

    float sum = v1 + v2; if (sum == 0.f) sum = 1.f;
    float w1 = v1 / sum, w2 = v2 / sum;

    int pos = atomicAdd(&count[i1], 1);
    tok_idx[i1 * TNUM + pos] = t; tok_w[i1 * TNUM + pos] = w1;
    pos = atomicAdd(&count[i2], 1);
    tok_idx[i2 * TNUM + pos] = t; tok_w[i2 * TNUM + pos] = w2;
  }
}

// ---------------------------------------------------------------------------
// Kernel 2: grouped expert SwiGLU with bf16 WMMA, fp32 accumulate.
// TM=32 tokens/block: two 16-row A tiles share every B operand (2x WMMA per
// weight byte). Grid = ENUM * (TNUM/TM); blocks beyond count[e] exit early.
// PREBF=true: bf16 weights/activations from ws; gather uses CDNA5 async
// global->LDS copies. PREBF=false: inline fp32->bf16 conversion fallback.
// ---------------------------------------------------------------------------
template <bool PREBF>
__global__ __launch_bounds__(256)
void moe_expert_kernel(const float* __restrict__ x,
                       const unsigned short* __restrict__ x_b,
                       const float* __restrict__ gate_f,
                       const float* __restrict__ up_f,
                       const float* __restrict__ down_f,
                       const unsigned short* __restrict__ gate_b,
                       const unsigned short* __restrict__ up_b,
                       const unsigned short* __restrict__ down_b,
                       const int*   __restrict__ count,
                       const int*   __restrict__ tok_idx,
                       const float* __restrict__ tok_w,
                       float* __restrict__ out) {
  const int e    = blockIdx.x >> 6;      // TNUM/TM = 64 tiles per expert
  const int tile = blockIdx.x & 63;
  const int cnt  = count[e];
  const int row0 = tile * TM;
  if (row0 >= cnt) return;
  const int nrows = min(TM, cnt - row0);

  __shared__ unsigned short sX[TM][HDIM];   // 32 x 2048 bf16 = 128 KB
  __shared__ unsigned short sH[TM][IDIM];   // 32 x  768 bf16 =  48 KB
  __shared__ int   sTok[TM];
  __shared__ float sWt[TM];

  const int tid = threadIdx.x;
  if (tid < TM) {
    int g = row0 + tid;
    sTok[tid] = (g < cnt) ? tok_idx[e * TNUM + g] : 0;
    sWt[tid]  = (g < cnt) ? tok_w[e * TNUM + g] : 0.f;
  }
  __syncthreads();

  // ---- gather token rows -> LDS as bf16 (rows past nrows are zero) ----
  for (int idx = tid; idx < TM * (HDIM / 8); idx += 256) {
    int r = idx >> 8;               // HDIM/8 = 256 chunks of 8 elems per row
    int c = (idx & 255) * 8;
    if (r < nrows) {
      if (PREBF) {
        // pure byte copy: async DMA straight into LDS (ASYNCcnt-tracked)
        async_copy_b128((unsigned)(unsigned long long)&sX[r][c],
                        x_b + (size_t)sTok[r] * HDIM + c);
      } else {
        *(v8us*)&sX[r][c] = cvt8(&x[(size_t)sTok[r] * HDIM + c]);
      }
    } else {
      v8us z = {0, 0, 0, 0, 0, 0, 0, 0};
      *(v8us*)&sX[r][c] = z;
    }
  }
  if (PREBF) wait_asynccnt0();
  __syncthreads();

  const int ln    = tid & 15;                // M (A) or N (B/C) within tile
  const int koff  = (tid & 16) ? 8 : 0;      // unused: A and B both take their K half from kb
  const int kb    = (tid & 16) ? 16 : 0;     // B operand K-half select
  const int wv    = tid >> 5;
  const int rbase = (tid & 16) >> 1;         // C/D rows: 0..7 or 8..15

  // ---- stage A: G = X*gate^T, U = X*up^T ; h = silu(G)*U -> LDS ----
  for (int nb = wv; nb < IDIM / 16; nb += 8) {
    v8f accG0 = {}, accG1 = {};
    v8f accU0 = {}, accU1 = {};
    const size_t wrow = ((size_t)e * IDIM + nb * 16 + ln) * HDIM + kb;
    const unsigned short* gb = gate_b + wrow;
    const unsigned short* ub = up_b   + wrow;
    const float*          gf = gate_f + wrow;
    const float*          uf = up_f   + wrow;
#pragma unroll 2
    for (int k0 = 0; k0 < HDIM; k0 += 32) {
      v16bf a0 = mk_lds(&sX[ln     ][k0 + kb], &sX[ln     ][k0 + kb + 8]);
      v16bf a1 = mk_lds(&sX[ln + 16][k0 + kb], &sX[ln + 16][k0 + kb + 8]);
      v16bf bg = PREBF ? ld16bf(gb + k0) : cvt16(gf + k0);
      v16bf bu = PREBF ? ld16bf(ub + k0) : cvt16(uf + k0);
      accG0 = wmma_bf16(a0, bg, accG0);
      accG1 = wmma_bf16(a1, bg, accG1);
      accU0 = wmma_bf16(a0, bu, accU0);
      accU1 = wmma_bf16(a1, bu, accU1);
    }
    const int col = nb * 16 + ln;
#pragma unroll
    for (int j = 0; j < 8; j++) {
      float g0 = accG0[j], u0 = accU0[j];
      float g1 = accG1[j], u1 = accU1[j];
      // silu(g)*u = g*u * rcp(1 + exp(-g))  (fast rcp, no IEEE divide)
      float h0 = g0 * u0 * __builtin_amdgcn_rcpf(1.f + __expf(-g0));
      float h1 = g1 * u1 * __builtin_amdgcn_rcpf(1.f + __expf(-g1));
      sH[rbase + j     ][col] = f2bf(h0);
      sH[rbase + j + 16][col] = f2bf(h1);
    }
  }
  __syncthreads();

  // ---- stage B: Y = h * down^T ; out[tok] += w * Y (atomic f32) ----
  for (int nb = wv; nb < HDIM / 16; nb += 8) {
    v8f acc0 = {}, acc1 = {};
    const size_t wrow = ((size_t)e * HDIM + nb * 16 + ln) * IDIM + kb;
    const unsigned short* db = down_b + wrow;
    const float*          df = down_f + wrow;
#pragma unroll 4
    for (int k0 = 0; k0 < IDIM; k0 += 32) {
      v16bf a0 = mk_lds(&sH[ln     ][k0 + kb], &sH[ln     ][k0 + kb + 8]);
      v16bf a1 = mk_lds(&sH[ln + 16][k0 + kb], &sH[ln + 16][k0 + kb + 8]);
      v16bf b  = PREBF ? ld16bf(db + k0) : cvt16(df + k0);
      acc0 = wmma_bf16(a0, b, acc0);
      acc1 = wmma_bf16(a1, b, acc1);
    }
    const int col = nb * 16 + ln;
#pragma unroll
    for (int j = 0; j < 8; j++) {
      int r0 = rbase + j;
      int r1 = r0 + 16;
      if (r0 < nrows)
        atomicAdd(&out[(size_t)sTok[r0] * HDIM + col], acc0[j] * sWt[r0]);
      if (r1 < nrows)
        atomicAdd(&out[(size_t)sTok[r1] * HDIM + col], acc1[j] * sWt[r1]);
    }
  }
}

// ---------------------------------------------------------------------------
extern "C" void kernel_launch(void* const* d_in, const int* in_sizes, int n_in,
                              void* d_out, int out_size, void* d_ws, size_t ws_size,
                              hipStream_t stream) {
  (void)in_sizes; (void)n_in;
  const float* x  = (const float*)d_in[0];   // [2,1024,2048]
  const float* rw = (const float*)d_in[1];   // [8,2048]
  const float* gw = (const float*)d_in[2];   // [8,768,2048]
  const float* uw = (const float*)d_in[3];   // [8,768,2048]
  const float* dw = (const float*)d_in[4];   // [8,2048,768]
  float* out = (float*)d_out;                // [2,1024,2048]

  // workspace layout (re-initialized every launch; no cross-call state)
  int*   count   = (int*)d_ws;                       // [16] (8 used)
  int*   tok_idx = count + 16;                       // [E * T]
  float* tok_w   = (float*)(tok_idx + ENUM * TNUM);  // [E * T]
  unsigned short* gate_b = (unsigned short*)(tok_w + ENUM * TNUM);
  const size_t wcnt = (size_t)ENUM * IDIM * HDIM;    // elems per weight matrix
  const size_t xcnt = (size_t)TNUM * HDIM;           // activation elems
  unsigned short* up_b   = gate_b + wcnt;
  unsigned short* down_b = up_b + wcnt;
  unsigned short* x_b    = down_b + wcnt;
  const size_t need = (size_t)((char*)(x_b + xcnt) - (char*)d_ws);
  const bool prebf = (ws_size >= need);

  zero_kernel<<<512, 256, 0, stream>>>((float4*)out, out_size / 4, count);
  router_kernel<<<TNUM, 256, 0, stream>>>(x, rw, count, tok_idx, tok_w);

  if (prebf) {
    cvt_kernel<<<2048, 256, 0, stream>>>(gw, gate_b, (int)(wcnt / 8));
    cvt_kernel<<<2048, 256, 0, stream>>>(uw, up_b,   (int)(wcnt / 8));
    cvt_kernel<<<2048, 256, 0, stream>>>(dw, down_b, (int)(wcnt / 8));
    cvt_kernel<<<1024, 256, 0, stream>>>(x,  x_b,    (int)(xcnt / 8));
    moe_expert_kernel<true><<<ENUM * (TNUM / TM), 256, 0, stream>>>(
        x, x_b, gw, uw, dw, gate_b, up_b, down_b, count, tok_idx, tok_w, out);
  } else {
    moe_expert_kernel<false><<<ENUM * (TNUM / TM), 256, 0, stream>>>(
        x, x_b, gw, uw, dw, gate_b, up_b, down_b, count, tok_idx, tok_w, out);
  }
}
